// MultiheadSelfAttention_67138928771075
// MI455X (gfx1250) — matrix operand order repaired and checked by reading, not compiled, not run on hardware
//
#include <hip/hip_runtime.h>

// Problem constants (from the reference).
#define D_MODEL   2048
#define N_HEADS_C 16
#define N_KV      4
#define HD        128
#define KV_DIM_C  512
#define QKV_N     3072            // D_MODEL + 2*KV_DIM
#define SEQ       2048
#define BATCH     4
#define NTOK      (BATCH * SEQ)   // 8192
#define EPS_F     1e-6f

typedef __attribute__((ext_vector_type(16))) __bf16 v16bf;
typedef __attribute__((ext_vector_type(8)))  __bf16 v8bf;
typedef __attribute__((ext_vector_type(4)))  __bf16 v4bf;
typedef __attribute__((ext_vector_type(8)))  float  v8f;

union Frag { v16bf v; v8bf h[2]; };

__device__ __forceinline__ v8f wmma_bf16(v16bf a, v16bf b, v8f c) {
  // D = A(16x32 bf16) x B(32x16 bf16) + C(16x16 f32)
  return __builtin_amdgcn_wmma_f32_16x16x32_bf16(false, a, false, b, (short)0, c,
                                                 false, false);
}

__device__ __forceinline__ v8f vzero8() {
  v8f z;
#pragma unroll
  for (int i = 0; i < 8; ++i) z[i] = 0.0f;
  return z;
}

// ---------------------------------------------------------------------------
// Elementwise f32 -> bf16 (for x).  8 elements / thread.
// ---------------------------------------------------------------------------
__global__ __launch_bounds__(256)
void to_bf16(const float* __restrict__ src, __bf16* __restrict__ dst) {
  const size_t i = ((size_t)blockIdx.x * 256 + threadIdx.x) * 8;
  float4 a = *reinterpret_cast<const float4*>(src + i);
  float4 b = *reinterpret_cast<const float4*>(src + i + 4);
  v8bf o;
  o[0] = (__bf16)a.x; o[1] = (__bf16)a.y; o[2] = (__bf16)a.z; o[3] = (__bf16)a.w;
  o[4] = (__bf16)b.x; o[5] = (__bf16)b.y; o[6] = (__bf16)b.z; o[7] = (__bf16)b.w;
  *reinterpret_cast<v8bf*>(dst + i) = o;
}

// ---------------------------------------------------------------------------
// W[K][N] f32  ->  Wt[N][K] bf16  (32x32 tiles through LDS).
// ---------------------------------------------------------------------------
__global__ __launch_bounds__(256)
void transpose_to_bf16(const float* __restrict__ W, __bf16* __restrict__ Wt,
                       int K, int N) {
  __shared__ __align__(16) __bf16 t[32 * 40];
  const int tid = threadIdx.x;
  const int k0 = blockIdx.x * 32;
  const int n0 = blockIdx.y * 32;
  {
    const int kr = tid >> 3;
    const int nq = (tid & 7) << 2;
    float4 f = *reinterpret_cast<const float4*>(W + (size_t)(k0 + kr) * N + n0 + nq);
    t[(nq + 0) * 40 + kr] = (__bf16)f.x;
    t[(nq + 1) * 40 + kr] = (__bf16)f.y;
    t[(nq + 2) * 40 + kr] = (__bf16)f.z;
    t[(nq + 3) * 40 + kr] = (__bf16)f.w;
  }
  __syncthreads();
  if (tid < 128) {
    const int nr = tid >> 2;
    const int q  = (tid & 3) << 3;
    v8bf o = *reinterpret_cast<const v8bf*>(t + nr * 40 + q);
    *reinterpret_cast<v8bf*>(Wt + (size_t)(n0 + nr) * K + k0 + q) = o;
  }
}

// ---------------------------------------------------------------------------
// C[M,N] = A[M,K] @ Bt[N,K]^T + bias.  A/Bt bf16 in memory, f32 accum/out.
// Block: 256 thr (8 waves), tile 128x128, wave tile 64x32, K-step 32.
// Double-buffered LDS staged with GLOBAL_LOAD_ASYNC_TO_LDS_B128 (ASYNCcnt),
// prefetching tile i+1 while tile i feeds the WMMAs.
// LDS tiles: [row][32 k] bf16, row stride 40 (pad) -> fragments are two
// contiguous ds_load_b128 each.
// ---------------------------------------------------------------------------
#define TILE_ELEMS (128 * 40)
#define TILE_BYTES (TILE_ELEMS * 2)

__global__ __launch_bounds__(256, 2)
void gemm_bf16(const __bf16* __restrict__ A, const __bf16* __restrict__ Bt,
               const float* __restrict__ bias, float* __restrict__ C,
               int M, int N, int K) {
  __shared__ __align__(16) __bf16 lsA[2][TILE_ELEMS];
  __shared__ __align__(16) __bf16 lsB[2][TILE_ELEMS];

  const int tid  = threadIdx.x;
  const int lane = tid & 31;
  const int wid  = tid >> 5;
  const int g    = lane >> 4;   // 16-lane half
  const int ln   = lane & 15;
  const int wm   = wid >> 2;    // 0..1 (64-row strip)
  const int wn   = wid & 3;     // 0..3 (32-col strip)
  const int m0   = blockIdx.x * 128;
  const int n0   = blockIdx.y * 128;

  const __bf16* Abase = A  + (size_t)m0 * K;
  const __bf16* Bbase = Bt + (size_t)n0 * K;
  const unsigned ldsA0 = (unsigned)(unsigned long long)(&lsA[0][0]);
  const unsigned ldsB0 = (unsigned)(unsigned long long)(&lsB[0][0]);

  // Per-thread staging coords: 512 x 16B segments per tile, 2 per thread.
  const int rowS0 = tid >> 2,          qS0 = tid & 3;
  const int rowS1 = (tid + 256) >> 2,  qS1 = tid & 3;

  auto stage = [&](int buf, int kb) {
#pragma unroll
    for (int s = 0; s < 2; ++s) {
      const int row = s ? rowS1 : rowS0;
      const int q   = s ? qS1   : qS0;
      const unsigned lofs = (unsigned)(buf * TILE_BYTES + row * 80 + q * 16);
      const int      gofs = (row * K + kb + q * 8) * 2;  // bytes
      const unsigned la = ldsA0 + lofs;
      const unsigned lb = ldsB0 + lofs;
      asm volatile("global_load_async_to_lds_b128 %0, %1, %2"
                   :: "v"(la), "v"(gofs), "s"(Abase) : "memory");
      asm volatile("global_load_async_to_lds_b128 %0, %1, %2"
                   :: "v"(lb), "v"(gofs), "s"(Bbase) : "memory");
    }
  };

  v8f acc[4][2];
#pragma unroll
  for (int mt = 0; mt < 4; ++mt)
#pragma unroll
    for (int nt = 0; nt < 2; ++nt) acc[mt][nt] = vzero8();

  const int nk = K >> 5;
  stage(0, 0);

  for (int it = 0; it < nk; ++it) {
    asm volatile("s_wait_asynccnt 0x0" ::: "memory");  // tile `it` landed
    __syncthreads();
    if (it + 1 < nk) stage((it + 1) & 1, (it + 1) << 5);  // prefetch next

    const __bf16* tA = lsA[it & 1];
    const __bf16* tB = lsB[it & 1];

    // B fragments: lane = column n=ln, K = g*8+j | 16+g*8+j.
    Frag bfr[2];
#pragma unroll
    for (int nt = 0; nt < 2; ++nt) {
      const __bf16* p = tB + (wn * 32 + nt * 16 + ln) * 40 + g * 8;
      bfr[nt].h[0] = *reinterpret_cast<const v8bf*>(p);
      bfr[nt].h[1] = *reinterpret_cast<const v8bf*>(p + 16);
    }
    // A fragments: lane = row m=ln; K = g*8+j | 16+g*8+j.
#pragma unroll
    for (int mt = 0; mt < 4; ++mt) {
      const __bf16* p = tA + (wm * 64 + mt * 16 + ln) * 40 + g * 8;
      Frag afr;
      afr.h[0] = *reinterpret_cast<const v8bf*>(p);
      afr.h[1] = *reinterpret_cast<const v8bf*>(p + 16);
#pragma unroll
      for (int nt = 0; nt < 2; ++nt)
        acc[mt][nt] = wmma_bf16(afr.v, bfr[nt].v, acc[mt][nt]);
    }
  }

  // Epilogue: C/D layout — M = v + 8*g, N = ln.
#pragma unroll
  for (int mt = 0; mt < 4; ++mt) {
    const int mm0 = m0 + wm * 64 + mt * 16 + g * 8;
#pragma unroll
    for (int nt = 0; nt < 2; ++nt) {
      const int nn = n0 + wn * 32 + nt * 16 + ln;
      float* cp = C + (size_t)mm0 * N + nn;
      const float bv = bias[nn];
#pragma unroll
      for (int v = 0; v < 8; ++v) cp[(size_t)v * N] = acc[mt][nt][v] + bv;
    }
  }
}

// ---------------------------------------------------------------------------
// RMSNorm q/k heads + bf16 packing.  One wave per (token, slot).
// Slots: 0..15 q -> qb[N,H,L,128]; 16..19 k -> kb[N,Hkv,L,128];
//        20..23 v -> vbt[N,Hkv,128,L] (dim-major for contiguous P@V B-frags).
// ---------------------------------------------------------------------------
__global__ __launch_bounds__(256)
void rmsnorm_pack(const float* __restrict__ qkv, const float* __restrict__ q_scale,
                  const float* __restrict__ k_scale,
                  __bf16* __restrict__ qb, __bf16* __restrict__ kb,
                  __bf16* __restrict__ vbt) {
  const int gw   = blockIdx.x * 8 + (threadIdx.x >> 5);
  const int lane = threadIdx.x & 31;
  const int tok  = gw / 24;
  const int slot = gw - tok * 24;
  const int n    = tok / SEQ;
  const int l    = tok - n * SEQ;

  int col0;
  if (slot < 16)      col0 = slot * HD;
  else if (slot < 20) col0 = D_MODEL + (slot - 16) * HD;
  else                col0 = D_MODEL + KV_DIM_C + (slot - 20) * HD;

  const float4 f =
      *reinterpret_cast<const float4*>(qkv + (size_t)tok * QKV_N + col0 + lane * 4);

  if (slot < 20) {
    float ss = f.x * f.x + f.y * f.y + f.z * f.z + f.w * f.w;
#pragma unroll
    for (int off = 16; off > 0; off >>= 1) ss += __shfl_xor(ss, off, 32);
    const float r = rsqrtf(ss * (1.0f / (float)HD) + EPS_F);
    const float4 sc =
        *reinterpret_cast<const float4*>(((slot < 16) ? q_scale : k_scale) + lane * 4);
    v4bf o;
    o[0] = (__bf16)(f.x * sc.x * r);
    o[1] = (__bf16)(f.y * sc.y * r);
    o[2] = (__bf16)(f.z * sc.z * r);
    o[3] = (__bf16)(f.w * sc.w * r);
    __bf16* dst;
    if (slot < 16)
      dst = qb + ((size_t)(n * N_HEADS_C + slot) * SEQ + l) * HD + lane * 4;
    else
      dst = kb + ((size_t)(n * N_KV + (slot - 16)) * SEQ + l) * HD + lane * 4;
    *reinterpret_cast<v4bf*>(dst) = o;
  } else {
    const int h = slot - 20;
    __bf16* base = vbt + ((size_t)(n * N_KV + h) * HD + lane * 4) * SEQ + l;
    float fv[4] = {f.x, f.y, f.z, f.w};
#pragma unroll
    for (int c = 0; c < 4; ++c) base[(size_t)c * SEQ] = (__bf16)fv[c];
  }
}

// ---------------------------------------------------------------------------
// Causal flash attention.  Grid: (N*H, L/128).  8 waves/block; each wave owns
// 16 query rows, key blocks of 32, online softmax, all matmuls via WMMA.
// Output written directly as bf16 (feeds the bf16 out-projection).
// ---------------------------------------------------------------------------
__global__ __launch_bounds__(256)
void attn_fwd(const __bf16* __restrict__ qb, const __bf16* __restrict__ kbuf,
              const __bf16* __restrict__ vbt, __bf16* __restrict__ y) {
  __shared__ __align__(16) __bf16 pls[8 * 16 * 40];  // per-wave 16x32 P strip

  const int lane = threadIdx.x & 31;
  const int wid  = threadIdx.x >> 5;
  const int g    = lane >> 4;
  const int ln   = lane & 15;
  const int nh   = blockIdx.x;
  const int n    = nh >> 4;
  const int h    = nh & 15;
  const int kvh  = h >> 2;  // GQA: n_rep = 4
  const int qr0  = blockIdx.y * 128 + wid * 16;

  const __bf16* Q  = qb   + (size_t)(n * N_HEADS_C + h) * SEQ * HD;
  const __bf16* Kh = kbuf + (size_t)(n * N_KV + kvh) * SEQ * HD;
  const __bf16* Vt = vbt  + (size_t)(n * N_KV + kvh) * HD * SEQ;
  __bf16* pw = pls + wid * (16 * 40);

  // Q fragments (16 rows x 128 dims = 4 A-fragments of K=32).
  Frag aq[4];
  {
    const __bf16* qrow = Q + (size_t)(qr0 + ln) * HD;
#pragma unroll
    for (int c = 0; c < 4; ++c) {
      aq[c].h[0] = *reinterpret_cast<const v8bf*>(qrow + c * 32 + g * 8);
      aq[c].h[1] = *reinterpret_cast<const v8bf*>(qrow + c * 32 + g * 8 + 16);
    }
  }

  v8f o[8];
#pragma unroll
  for (int t = 0; t < 8; ++t) o[t] = vzero8();
  float mrow[8], lrow[8];
#pragma unroll
  for (int v = 0; v < 8; ++v) { mrow[v] = -1e30f; lrow[v] = 0.0f; }

  const float sc = 0.08838834764831845f;  // 1/sqrt(128)
  const int nkb  = (qr0 + 47) >> 5;       // key blocks through the diagonal

  for (int kblk = 0; kblk < nkb; ++kblk) {
    const int k0 = kblk * 32;

    // S = Q @ K^T : two 16x16 tiles (keys k0..+15, k0+16..+31).
    v8f s0 = vzero8(), s1 = vzero8();
#pragma unroll
    for (int c = 0; c < 4; ++c) {
      const __bf16* kr0 = Kh + (size_t)(k0 + ln) * HD + c * 32 + g * 8;
      Frag bk0;
      bk0.h[0] = *reinterpret_cast<const v8bf*>(kr0);
      bk0.h[1] = *reinterpret_cast<const v8bf*>(kr0 + 16);
      s0 = wmma_bf16(aq[c].v, bk0.v, s0);
      const __bf16* kr1 = Kh + (size_t)(k0 + 16 + ln) * HD + c * 32 + g * 8;
      Frag bk1;
      bk1.h[0] = *reinterpret_cast<const v8bf*>(kr1);
      bk1.h[1] = *reinterpret_cast<const v8bf*>(kr1 + 16);
      s1 = wmma_bf16(aq[c].v, bk1.v, s1);
    }

    // Scale, causal mask, online softmax (row = v + 8*g, key col = ln / 16+ln).
    float alpha[8];
#pragma unroll
    for (int v = 0; v < 8; ++v) {
      const int q_abs = qr0 + v + g * 8;
      float e0 = (k0 + ln      <= q_abs) ? s0[v] * sc : -1e30f;
      float e1 = (k0 + 16 + ln <= q_abs) ? s1[v] * sc : -1e30f;
      float mx = fmaxf(e0, e1);
      mx = fmaxf(mx, __shfl_xor(mx, 1, 32));
      mx = fmaxf(mx, __shfl_xor(mx, 2, 32));
      mx = fmaxf(mx, __shfl_xor(mx, 4, 32));
      mx = fmaxf(mx, __shfl_xor(mx, 8, 32));
      const float mnew = fmaxf(mrow[v], mx);
      const float a    = __expf(mrow[v] - mnew);
      mrow[v] = mnew;
      const float p0 = __expf(e0 - mnew);
      const float p1 = __expf(e1 - mnew);
      pw[(v + g * 8) * 40 + ln]      = (__bf16)p0;  // C-layout -> LDS
      pw[(v + g * 8) * 40 + 16 + ln] = (__bf16)p1;
      float rs = p0 + p1;
      rs += __shfl_xor(rs, 1, 32);
      rs += __shfl_xor(rs, 2, 32);
      rs += __shfl_xor(rs, 4, 32);
      rs += __shfl_xor(rs, 8, 32);
      lrow[v]  = lrow[v] * a + rs;
      alpha[v] = a;
    }
#pragma unroll
    for (int t = 0; t < 8; ++t)
#pragma unroll
      for (int v = 0; v < 8; ++v) o[t][v] *= alpha[v];

    // Reload P as an A-fragment (16x32) from the wave-private LDS strip.
    Frag ap;
    {
      const __bf16* p = pw + ln * 40 + g * 8;
      ap.h[0] = *reinterpret_cast<const v8bf*>(p);
      ap.h[1] = *reinterpret_cast<const v8bf*>(p + 16);
    }

    // O += P @ V : V is dim-major, so B-fragments are contiguous in keys.
#pragma unroll
    for (int nt = 0; nt < 8; ++nt) {
      const __bf16* vp = Vt + (size_t)(nt * 16 + ln) * SEQ + k0 + g * 8;
      Frag bv;
      bv.h[0] = *reinterpret_cast<const v8bf*>(vp);
      bv.h[1] = *reinterpret_cast<const v8bf*>(vp + 16);
      o[nt] = wmma_bf16(ap.v, bv.v, o[nt]);
    }
  }

  float invl[8];
#pragma unroll
  for (int v = 0; v < 8; ++v) invl[v] = 1.0f / lrow[v];

#pragma unroll
  for (int nt = 0; nt < 8; ++nt) {
#pragma unroll
    for (int v = 0; v < 8; ++v) {
      const int q_abs = qr0 + v + g * 8;
      y[((size_t)n * SEQ + q_abs) * D_MODEL + h * HD + nt * 16 + ln] =
          (__bf16)(o[nt][v] * invl[v]);
    }
  }
}

// ---------------------------------------------------------------------------
extern "C" void kernel_launch(void* const* d_in, const int* in_sizes, int n_in,
                              void* d_out, int out_size, void* d_ws, size_t ws_size,
                              hipStream_t stream) {
  (void)in_sizes; (void)n_in; (void)out_size; (void)ws_size;
  const float* x       = (const float*)d_in[0];
  const float* Wqkv    = (const float*)d_in[1];
  const float* bqkv    = (const float*)d_in[2];
  const float* q_scale = (const float*)d_in[3];
  const float* k_scale = (const float*)d_in[4];
  const float* Wout    = (const float*)d_in[5];
  const float* bout    = (const float*)d_in[6];
  float* out = (float*)d_out;

  char* ws = (char*)d_ws;
  size_t off = 0;
  float*  qkv   = (float*)(ws + off);  off += (size_t)NTOK * QKV_N * sizeof(float);   // 96 MB
  __bf16* xb    = (__bf16*)(ws + off); off += (size_t)NTOK * D_MODEL * 2;             // 32 MB (reused as y)
  __bf16* wqkvT = (__bf16*)(ws + off); off += (size_t)QKV_N * D_MODEL * 2;            // 12 MB
  __bf16* woutT = (__bf16*)(ws + off); off += (size_t)D_MODEL * D_MODEL * 2;          //  8 MB
  __bf16* qb    = (__bf16*)(ws + off); off += (size_t)BATCH * N_HEADS_C * SEQ * HD * 2;
  __bf16* kb    = (__bf16*)(ws + off); off += (size_t)BATCH * N_KV * SEQ * HD * 2;
  __bf16* vbt   = (__bf16*)(ws + off);
  __bf16* y     = xb;  // safe reuse: xb is dead after GEMM-1

  // 0) One-time precision/layout prep (all L2-resident afterwards).
  to_bf16<<<dim3(NTOK * D_MODEL / 2048), 256, 0, stream>>>(x, xb);
  transpose_to_bf16<<<dim3(D_MODEL / 32, QKV_N / 32), 256, 0, stream>>>(
      Wqkv, wqkvT, D_MODEL, QKV_N);
  transpose_to_bf16<<<dim3(D_MODEL / 32, D_MODEL / 32), 256, 0, stream>>>(
      Wout, woutT, D_MODEL, D_MODEL);

  // 1) QKV projection: [8192,2048] @ [2048,3072] + bias  (async-staged WMMA)
  gemm_bf16<<<dim3(NTOK / 128, QKV_N / 128), 256, 0, stream>>>(
      xb, wqkvT, bqkv, qkv, NTOK, QKV_N, D_MODEL);
  // 2) RMSNorm q/k + pack q/k/v to bf16 (v transposed dim-major)
  rmsnorm_pack<<<dim3(NTOK * 24 / 8), 256, 0, stream>>>(
      qkv, q_scale, k_scale, qb, kb, vbt);
  // 3) Causal flash attention (GQA 16 q-heads over 4 kv-heads)
  attn_fwd<<<dim3(BATCH * N_HEADS_C, SEQ / 128), 256, 0, stream>>>(qb, kb, vbt, y);
  // 4) Output projection: [8192,2048] @ [2048,2048] + bias
  gemm_bf16<<<dim3(NTOK / 128, D_MODEL / 128), 256, 0, stream>>>(
      y, woutT, bout, out, NTOK, D_MODEL, D_MODEL);
}
